// Scage_3607772529426
// MI455X (gfx1250) — matrix operand order repaired and checked by reading, not compiled, not run on hardware
//
#include <hip/hip_runtime.h>
#include <hip/hip_bf16.h>

typedef __attribute__((ext_vector_type(16))) __bf16 v16bf;
typedef __attribute__((ext_vector_type(8)))  __bf16 v8bf;
typedef __attribute__((ext_vector_type(8)))  float  v8f;

#define WMMA_BF16(a, b, c) \
  __builtin_amdgcn_wmma_f32_16x16x32_bf16(false, (a), false, (b), (short)0, (c), false, false)

#define CAT16(lo, hi) \
  __builtin_shufflevector((lo), (hi), 0, 1, 2, 3, 4, 5, 6, 7, 8, 9, 10, 11, 12, 13, 14, 15)

// ---------------- fragment loaders (CDNA5 ISA 7.12.2 layouts) ----------------
// A-matrix 16x32 bf16, storage row-major with row stride 32 elements.
// Per-lane data is two contiguous 16B runs: [m, hi8..hi8+7] and [m, 16+hi8..23+hi8].
__device__ __forceinline__ v16bf frag_a32(const __bf16* p, int lane) {
  int m = lane & 15;
  int hi8 = (lane & 16) ? 8 : 0;
  const __bf16* r = p + m * 32 + hi8;
  v8bf lo = *(const v8bf*)(r);
  v8bf hi = *(const v8bf*)(r + 16);
  return CAT16(lo, hi);
}

// B-matrix 32x16 bf16 from lane-major storage p[n*ldn + k]:
// lane n holds K = hi8..hi8+7 and 16+hi8..23+hi8 -> two ds_load_b128.
__device__ __forceinline__ v16bf frag_b_lm(const __bf16* p, int ldn, int lane) {
  int n = lane & 15;
  int koff = (lane & 16) ? 8 : 0;
  const __bf16* r = p + n * ldn + koff;
  v8bf lo = *(const v8bf*)(r);
  v8bf hi = *(const v8bf*)(r + 16);
  return CAT16(lo, hi);
}

// ---------------- fp32 -> bf16 conversion ----------------
__global__ void cvt_f2b(const float* __restrict__ s, __bf16* __restrict__ d, int n) {
  int i = blockIdx.x * blockDim.x + threadIdx.x;
  if (i < n) d[i] = (__bf16)s[i];
}

// fp32 [K][N] -> bf16 transposed [N][K]  (one-time; keeps GEMM LDS fills contiguous)
__global__ void cvt_f2b_t(const float* __restrict__ s, __bf16* __restrict__ d,
                          int K, int N) {
  int i = blockIdx.x * blockDim.x + threadIdx.x;
  if (i < K * N) {
    int k = i / N, n = i - k * N;
    d[(size_t)n * K + k] = (__bf16)s[i];
  }
}

// ---------------- generic tiled WMMA GEMM: C = act(A@W + bias) ----------------
// A: [M,Kd] bf16 row-major. Wt: [Nd,Kd] bf16 (pre-transposed weights).
// 128x64 block tile, 4 waves; each wave: 32 rows x 64 cols = 8 WMMAs / k-step.
// outT=1: write bf16 output transposed per 256-row batch: CbT[(row/256), col, row%256].
__global__ __launch_bounds__(128) void gemm_bf16_wmma(
    const __bf16* __restrict__ A, const __bf16* __restrict__ Wt,
    const float* __restrict__ bias,
    float* __restrict__ Cf, __bf16* __restrict__ Cb,
    int M, int Kd, int Nd, int act, int outT) {
  __shared__ alignas(16) __bf16 As[128 * 32];   // [row][k] lane-major
  __shared__ alignas(16) __bf16 Bs[64 * 32];    // [n][k]  lane-major
  int tid = threadIdx.x;
  int lane = tid & 31, wv = tid >> 5;
  int row0 = blockIdx.y * 128;
  int col0 = blockIdx.x * 64;

  v8f acc[2][4];
#pragma unroll
  for (int r = 0; r < 2; ++r)
#pragma unroll
    for (int t = 0; t < 4; ++t)
#pragma unroll
      for (int z = 0; z < 8; ++z) acc[r][t][z] = 0.f;

  int bn = tid >> 1, bh = (tid & 1) * 16;   // Bs fill: one 16-elem chunk/thread

  for (int k0 = 0; k0 < Kd; k0 += 32) {
    // A tile 128x32: 256 chunks of 16 elems, 2 per thread
#pragma unroll
    for (int c = 0; c < 2; ++c) {
      int idx = tid + c * 128;
      int r = idx >> 1, half = (idx & 1) * 16;
      const __bf16* src = A + (size_t)(row0 + r) * Kd + k0 + half;
      *(v8bf*)(As + r * 32 + half)     = *(const v8bf*)(src);
      *(v8bf*)(As + r * 32 + half + 8) = *(const v8bf*)(src + 8);
    }
    // B tile 64x32 from pre-transposed weights: contiguous copies
    {
      const __bf16* src = Wt + (size_t)(col0 + bn) * Kd + k0 + bh;
      *(v8bf*)(Bs + bn * 32 + bh)     = *(const v8bf*)(src);
      *(v8bf*)(Bs + bn * 32 + bh + 8) = *(const v8bf*)(src + 8);
    }
    if (k0 + 32 < Kd) {  // gfx1250 global_prefetch_b8 of next K tile
      __builtin_prefetch(A + (size_t)(row0 + (tid >> 1)) * Kd + k0 + 32, 0, 3);
      __builtin_prefetch(Wt + (size_t)(col0 + bn) * Kd + k0 + 32 + bh, 0, 3);
    }
    __syncthreads();
    v16bf a0 = frag_a32(&As[(wv * 32) * 32], lane);
    v16bf a1 = frag_a32(&As[(wv * 32 + 16) * 32], lane);
    v16bf bf0 = frag_b_lm(&Bs[0 * 512], 32, lane);
    v16bf bf1 = frag_b_lm(&Bs[1 * 512], 32, lane);
    v16bf bf2 = frag_b_lm(&Bs[2 * 512], 32, lane);
    v16bf bf3 = frag_b_lm(&Bs[3 * 512], 32, lane);
    acc[0][0] = WMMA_BF16(a0, bf0, acc[0][0]);
    acc[0][1] = WMMA_BF16(a0, bf1, acc[0][1]);
    acc[0][2] = WMMA_BF16(a0, bf2, acc[0][2]);
    acc[0][3] = WMMA_BF16(a0, bf3, acc[0][3]);
    acc[1][0] = WMMA_BF16(a1, bf0, acc[1][0]);
    acc[1][1] = WMMA_BF16(a1, bf1, acc[1][1]);
    acc[1][2] = WMMA_BF16(a1, bf2, acc[1][2]);
    acc[1][3] = WMMA_BF16(a1, bf3, acc[1][3]);
    __syncthreads();
  }

  int hi8 = (lane & 16) ? 8 : 0;
  int cn = lane & 15;
#pragma unroll
  for (int r = 0; r < 2; ++r) {
#pragma unroll
    for (int t = 0; t < 4; ++t) {
      int col = col0 + t * 16 + cn;
      float bv = bias ? bias[col] : 0.f;
#pragma unroll
      for (int i = 0; i < 8; ++i) {
        int row = row0 + wv * 32 + r * 16 + i + hi8;
        float v = acc[r][t][i] + bv;
        if (act == 1) {                       // SiLU
          v = v / (1.f + __expf(-v));
        } else if (act == 2) {                // exact GELU
          v = 0.5f * v * (1.f + erff(v * 0.70710678118654752f));
        }
        if (Cf) Cf[(size_t)row * Nd + col] = v;
        if (Cb) {
          if (outT) Cb[((size_t)(row >> 8) * Nd + col) * 256 + (row & 255)] = (__bf16)v;
          else      Cb[(size_t)row * Nd + col] = (__bf16)v;
        }
      }
    }
  }
}

// ---------------- fused flash attention over (b, head, dist-scale) ----------------
// qb/kb: [B*N, D] bf16 ; vT: [B][D][N] bf16 (batch-transposed V projection).
// Output scattered into torch-faithful permuted msg2 layout:
// msg2[b, kh*4 + d/8, (d%8)*256 + n], kh = ks*H + h.
__global__ __launch_bounds__(256) void attn_kernel(
    const __bf16* __restrict__ qb, const __bf16* __restrict__ kb,
    const __bf16* __restrict__ vT,
    const float* __restrict__ attn_bias, const int* __restrict__ mask,
    const float* __restrict__ dist, const float* __restrict__ dist_bar,
    __bf16* __restrict__ msg2) {
  const int Nn = 256, Dd = 512, Hh = 16, Kk = 4, dk = 32;
  int blk = blockIdx.x;       // b*64 + h*4 + ks
  int b = blk >> 6;
  int h = (blk >> 2) & 15;
  int ks = blk & 3;

  __shared__ alignas(16) __bf16 Kt[256 * 32];     // [seq][dk]  (lane-major for S frags)
  __shared__ alignas(16) __bf16 Vt_t[32 * 256];   // [dk][seq]  (lane-major for PV frags)
  __shared__ alignas(16) __bf16 Pb[8][16 * 32];   // per-wave P strip, row stride 32

  int tid = threadIdx.x, lane = tid & 31, wv = tid >> 5;
  // K: contiguous 16B copies (512 chunks of 16 elems)
  for (int c = tid; c < 512; c += 256) {
    int n = c >> 1, half = (c & 1) * 16;
    const __bf16* src = kb + (size_t)(b * Nn + n) * Dd + h * dk + half;
    *(v8bf*)(Kt + n * 32 + half)     = *(const v8bf*)(src);
    *(v8bf*)(Kt + n * 32 + half + 8) = *(const v8bf*)(src + 8);
  }
  // V: already [d][n] per (b); contiguous 16B copies (512 chunks)
  for (int c = tid; c < 512; c += 256) {
    int d = c >> 4, sg = (c & 15) * 16;
    const __bf16* src = vT + ((size_t)b * Dd + h * dk + d) * 256 + sg;
    *(v8bf*)(Vt_t + d * 256 + sg)     = *(const v8bf*)(src);
    *(v8bf*)(Vt_t + d * 256 + sg + 8) = *(const v8bf*)(src + 8);
  }
  __syncthreads();

  float dbar = dist_bar[ks];
  const float scale = 0.17677669529663687f;  // 1/sqrt(32), folded into epilogue
  int m = lane & 15;
  int hi8 = (lane & 16) ? 8 : 0;
  int cn = lane & 15;

  for (int it = 0; it < 2; ++it) {
    int strip = wv + it * 8;
    int r0 = strip * 16;

    // Q fragment straight from global: two 16B loads per lane
    v16bf aq;
    {
      const __bf16* qr = qb + (size_t)(b * Nn + r0 + m) * Dd + h * dk + hi8;
      v8bf lo = *(const v8bf*)(qr);
      v8bf hi = *(const v8bf*)(qr + 16);
      aq = CAT16(lo, hi);
    }

    v8f o0, o1;
    float rmax[8], rsum[8];
#pragma unroll
    for (int z = 0; z < 8; ++z) { o0[z] = 0.f; o1[z] = 0.f; rmax[z] = -3.0e38f; rsum[z] = 0.f; }

    for (int cc = 0; cc < 8; ++cc) {
      int c0 = cc * 32;
      v16bf bk0 = frag_b_lm(&Kt[c0 * 32], 32, lane);
      v16bf bk1 = frag_b_lm(&Kt[(c0 + 16) * 32], 32, lane);
      v8f s0, s1;
#pragma unroll
      for (int z = 0; z < 8; ++z) { s0[z] = 0.f; s1[z] = 0.f; }
      s0 = WMMA_BF16(aq, bk0, s0);
      s1 = WMMA_BF16(aq, bk1, s1);

      float p0[8], p1[8];
#pragma unroll
      for (int i = 0; i < 8; ++i) {
        int row = r0 + i + hi8;
        int ca = c0 + cn, cb2 = c0 + 16 + cn;
        float v0 = s0[i] * scale + attn_bias[((size_t)(b * Kk + ks) * Nn + row) * Nn + ca];
        float v1 = s1[i] * scale + attn_bias[((size_t)(b * Kk + ks) * Nn + row) * Nn + cb2];
        if (mask[(size_t)(b * Nn + row) * Nn + ca] == 0) v0 = -1e12f;
        if (mask[(size_t)(b * Nn + row) * Nn + cb2] == 0) v1 = -1e12f;
        bool al0 = (row == 0) || (ca == 0) ||
                   (dist[((size_t)b * 255 + (row - 1)) * 255 + (ca - 1)] < dbar);
        bool al1 = (row == 0) || (cb2 == 0) ||
                   (dist[((size_t)b * 255 + (row - 1)) * 255 + (cb2 - 1)] < dbar);
        if (!al0) v0 = -1e12f;
        if (!al1) v1 = -1e12f;
        p0[i] = v0;
        p1[i] = v1;
      }

      // online softmax update (per row, butterfly across the 16-lane half)
#pragma unroll
      for (int i = 0; i < 8; ++i) {
        float mx = fmaxf(p0[i], p1[i]);
#pragma unroll
        for (int off = 1; off < 16; off <<= 1) mx = fmaxf(mx, __shfl_xor(mx, off, 32));
        float nm = fmaxf(rmax[i], mx);
        float sc = __expf(rmax[i] - nm);
        float e0 = __expf(p0[i] - nm);
        float e1 = __expf(p1[i] - nm);
        float sm = e0 + e1;
#pragma unroll
        for (int off = 1; off < 16; off <<= 1) sm += __shfl_xor(sm, off, 32);
        rsum[i] = rsum[i] * sc + sm;
        rmax[i] = nm;
        o0[i] *= sc;
        o1[i] *= sc;
        Pb[wv][(i + hi8) * 32 + cn]      = (__bf16)e0;
        Pb[wv][(i + hi8) * 32 + 16 + cn] = (__bf16)e1;
      }
      // wave-private LDS round-trip: make DS stores visible before A-frag reads
      asm volatile("s_wait_dscnt 0" ::: "memory");

      v16bf ap  = frag_a32(&Pb[wv][0], lane);
      v16bf bv0 = frag_b_lm(&Vt_t[c0], 256, lane);              // dk cols 0-15
      v16bf bv1 = frag_b_lm(&Vt_t[16 * 256 + c0], 256, lane);   // dk cols 16-31
      o0 = WMMA_BF16(ap, bv0, o0);
      o1 = WMMA_BF16(ap, bv1, o1);
    }

    int kh = ks * Hh + h;
#pragma unroll
    for (int i = 0; i < 8; ++i) {
      int row = r0 + i + hi8;
      float inv = 1.f / rsum[i];
      {
        int d = cn;
        int n2 = kh * 4 + (d >> 3), c2 = ((d & 7) << 8) + row;
        msg2[((size_t)b * 256 + n2) * 2048 + c2] = (__bf16)(o0[i] * inv);
      }
      {
        int d = 16 + cn;
        int n2 = kh * 4 + (d >> 3), c2 = ((d & 7) << 8) + row;
        msg2[((size_t)b * 256 + n2) * 2048 + c2] = (__bf16)(o1[i] * inv);
      }
    }
  }
}

// ---------------- residual + LayerNorm over D=512 ----------------
__global__ __launch_bounds__(256) void ln_kernel(
    const float* __restrict__ hin, const float* __restrict__ res,
    const float* __restrict__ g, const float* __restrict__ be,
    float* __restrict__ yf, __bf16* __restrict__ yb) {
  __shared__ float red[256];
  int row = blockIdx.x, tid = threadIdx.x;
  size_t base = (size_t)row * 512;
  float v0 = hin[base + tid] + res[base + tid];
  float v1 = hin[base + 256 + tid] + res[base + 256 + tid];
  red[tid] = v0 + v1;
  __syncthreads();
  for (int o = 128; o > 0; o >>= 1) {
    if (tid < o) red[tid] += red[tid + o];
    __syncthreads();
  }
  float mean = red[0] * (1.f / 512.f);
  __syncthreads();
  float d0 = v0 - mean, d1 = v1 - mean;
  red[tid] = d0 * d0 + d1 * d1;
  __syncthreads();
  for (int o = 128; o > 0; o >>= 1) {
    if (tid < o) red[tid] += red[tid + o];
    __syncthreads();
  }
  float inv = rsqrtf(red[0] * (1.f / 512.f) + 1e-6f);
  float r0 = d0 * inv * g[tid] + be[tid];
  float r1 = d1 * inv * g[tid + 256] + be[tid + 256];
  if (yf) { yf[base + tid] = r0; yf[base + 256 + tid] = r1; }
  if (yb) { yb[base + tid] = (__bf16)r0; yb[base + 256 + tid] = (__bf16)r1; }
}

// ---------------- host orchestration ----------------
extern "C" void kernel_launch(void* const* d_in, const int* in_sizes, int n_in,
                              void* d_out, int out_size, void* d_ws, size_t ws_size,
                              hipStream_t stream) {
  const int B = 8, N = 256, D = 512, FFN = 2048, KH = 2048;  // KH = K*H*dk
  const int M = B * N;  // 2048 rows

  const float* x    = (const float*)d_in[0];
  const float* dist = (const float*)d_in[1];
  const float* dbar = (const float*)d_in[2];
  const float* abias = (const float*)d_in[3];
  const int*   mask = (const int*)d_in[4];
  const float* Wq = (const float*)d_in[6];  const float* bq = (const float*)d_in[7];
  const float* Wk = (const float*)d_in[8];  const float* bk = (const float*)d_in[9];
  const float* Wv = (const float*)d_in[10]; const float* bv = (const float*)d_in[11];
  const float* W1 = (const float*)d_in[12]; const float* b1 = (const float*)d_in[13];
  const float* W2 = (const float*)d_in[14]; const float* b2 = (const float*)d_in[15];
  const float* g1 = (const float*)d_in[16]; const float* be1 = (const float*)d_in[17];
  const float* Wf1 = (const float*)d_in[18]; const float* bf1 = (const float*)d_in[19];
  const float* Wf2 = (const float*)d_in[20]; const float* bf2 = (const float*)d_in[21];
  const float* g2 = (const float*)d_in[22]; const float* be2 = (const float*)d_in[23];
  float* out = (float*)d_out;

  // workspace carve-up (256B aligned)
  size_t off = 0;
  auto carve = [&](size_t bytes) -> char* {
    char* p = (char*)d_ws + off;
    off = (off + bytes + 255) & ~(size_t)255;
    return p;
  };
  __bf16* xb   = (__bf16*)carve((size_t)M * D * 2);
  __bf16* Wqb  = (__bf16*)carve((size_t)D * D * 2);    // [N][K] transposed
  __bf16* Wkb  = (__bf16*)carve((size_t)D * D * 2);
  __bf16* Wvb  = (__bf16*)carve((size_t)D * D * 2);
  __bf16* W1b  = (__bf16*)carve((size_t)KH * D * 2);
  __bf16* W2b  = (__bf16*)carve((size_t)D * D * 2);
  __bf16* Wf1b = (__bf16*)carve((size_t)D * FFN * 2);
  __bf16* Wf2b = (__bf16*)carve((size_t)FFN * D * 2);
  __bf16* qbuf = (__bf16*)carve((size_t)M * D * 2);
  __bf16* kbuf = (__bf16*)carve((size_t)M * D * 2);
  __bf16* vTb  = (__bf16*)carve((size_t)M * D * 2);    // [B][D][N]
  __bf16* msg2 = (__bf16*)carve((size_t)M * KH * 2);
  __bf16* t1b  = (__bf16*)carve((size_t)M * D * 2);
  float*  hbuf = (float*)carve((size_t)M * D * 4);
  float*  ybuf = (float*)carve((size_t)M * D * 4);
  __bf16* ybb  = (__bf16*)carve((size_t)M * D * 2);
  __bf16* t2b  = (__bf16*)carve((size_t)M * FFN * 2);
  float*  fbuf = (float*)carve((size_t)M * D * 4);

  cvt_f2b<<<(M * D + 255) / 256, 256, 0, stream>>>(x, xb, M * D);
  auto cvtT = [&](const float* s, __bf16* d, int K, int Nc) {
    cvt_f2b_t<<<(K * Nc + 255) / 256, 256, 0, stream>>>(s, d, K, Nc);
  };
  cvtT(Wq, Wqb, D, D);   cvtT(Wk, Wkb, D, D);   cvtT(Wv, Wvb, D, D);
  cvtT(W1, W1b, KH, D);  cvtT(W2, W2b, D, D);
  cvtT(Wf1, Wf1b, D, FFN); cvtT(Wf2, Wf2b, FFN, D);

  dim3 blk(128);
  // QKV projections (bf16 out only; V written batch-transposed)
  gemm_bf16_wmma<<<dim3(D / 64, M / 128), blk, 0, stream>>>(xb, Wqb, bq, nullptr, qbuf, M, D, D, 0, 0);
  gemm_bf16_wmma<<<dim3(D / 64, M / 128), blk, 0, stream>>>(xb, Wkb, bk, nullptr, kbuf, M, D, D, 0, 0);
  gemm_bf16_wmma<<<dim3(D / 64, M / 128), blk, 0, stream>>>(xb, Wvb, bv, nullptr, vTb, M, D, D, 0, 1);

  // fused flash attention -> permuted msg2
  attn_kernel<<<B * 16 * 4, 256, 0, stream>>>(qbuf, kbuf, vTb, abias, mask, dist, dbar, msg2);

  // h = silu(msg2@W1+b1)@W2+b2 ; y = LN(h + x)
  gemm_bf16_wmma<<<dim3(D / 64, M / 128), blk, 0, stream>>>(msg2, W1b, b1, nullptr, t1b, M, KH, D, 1, 0);
  gemm_bf16_wmma<<<dim3(D / 64, M / 128), blk, 0, stream>>>(t1b, W2b, b2, hbuf, nullptr, M, D, D, 0, 0);
  ln_kernel<<<M, 256, 0, stream>>>(hbuf, x, g1, be1, ybuf, ybb);

  // f = gelu(y@Wf1+bf1)@Wf2+bf2 ; out = LN(f + y)
  gemm_bf16_wmma<<<dim3(FFN / 64, M / 128), blk, 0, stream>>>(ybb, Wf1b, bf1, nullptr, t2b, M, D, FFN, 2, 0);
  gemm_bf16_wmma<<<dim3(D / 64, M / 128), blk, 0, stream>>>(t2b, Wf2b, bf2, fbuf, nullptr, M, FFN, D, 0, 0);
  ln_kernel<<<M, 256, 0, stream>>>(fbuf, ybuf, g2, be2, out, nullptr);

  (void)in_sizes; (void)n_in; (void)out_size; (void)ws_size;
}
